// SelfAttentionHead_23699629540193
// MI455X (gfx1250) — compile-verified, matrix operand order checked by reading, not run on hardware
//
#include <hip/hip_runtime.h>
#include <hip/hip_bf16.h>
#include <stdint.h>

// Problem constants (match reference)
#define Bn 8
#define Tn 2048
#define Cn 1024
#define Hn 64

typedef __attribute__((ext_vector_type(16))) _Float16 v16h;
typedef __attribute__((ext_vector_type(8)))  float    v8f;
typedef __attribute__((ext_vector_type(4)))  unsigned u32x4;
typedef __attribute__((ext_vector_type(8)))  int      i32x8;
typedef __attribute__((ext_vector_type(4)))  int      i32x4;

#if __has_builtin(__builtin_amdgcn_tensor_load_to_lds)
  #define HAVE_TDM 1
  #if __has_include(<hip/amd_detail/amd_gfx1250_TDM.h>)
    #define TDM_ARITY 6
  #else
    #define TDM_ARITY 5
  #endif
#else
  #define HAVE_TDM 0
#endif

template <int N>
__device__ __forceinline__ void wait_tensorcnt() {
#if __has_builtin(__builtin_amdgcn_s_wait_tensorcnt)
  __builtin_amdgcn_s_wait_tensorcnt((short)N);
#else
  asm volatile("s_wait_tensorcnt %0" :: "i"(N) : "memory");
#endif
}

__device__ __forceinline__ void wait_dscnt0() {
#if __has_builtin(__builtin_amdgcn_s_wait_dscnt)
  __builtin_amdgcn_s_wait_dscnt(0);
#else
  asm volatile("s_wait_dscnt 0x0" ::: "memory");
#endif
}

// Strided-2D tile load, Global -> LDS, via the Tensor Data Mover.
// Descriptor bit layout per CDNA5 ISA ch.8 (D# group0/group1).
// width/rows/stride in elements of size (1 << dsz_log2) bytes; the tile is
// packed densely into LDS ([rows][width]).
__device__ __forceinline__ void tdm_load_2d(void* lds_dst, const void* gsrc,
                                            unsigned dsz_log2, unsigned width,
                                            unsigned rows, unsigned stride,
                                            int lane) {
#if HAVE_TDM
  unsigned lds_off = (unsigned)(uintptr_t)lds_dst;       // low 32 bits = LDS byte offset
  unsigned long long ga = (unsigned long long)(uintptr_t)gsrc;
  u32x4 g0;
  g0[0] = 1u;                                            // count=1 (valid), user mode
  g0[1] = lds_off;                                       // lds_addr
  g0[2] = (unsigned)ga;                                  // global_addr[31:0]
  g0[3] = (unsigned)((ga >> 32) & 0x1FFFFFFull) | (2u << 30); // addr[56:32] | type=2
  i32x8 g1;
  g1[0] = (int)(dsz_log2 << 16);                         // wg_mask=0, data_size
  g1[1] = (int)((width & 0xFFFFu) << 16);                // tensor_dim0[15:0]
  g1[2] = (int)(((width >> 16) & 0xFFFFu) | ((rows & 0xFFFFu) << 16));
  g1[3] = (int)(((rows >> 16) & 0xFFFFu) | ((width & 0xFFFFu) << 16));  // | tile_dim0
  g1[4] = (int)(rows & 0xFFFFu);                         // tile_dim1 (tile_dim2=0)
  g1[5] = (int)stride;                                   // tensor_dim0_stride[31:0]
  g1[6] = 0;
  g1[7] = 0;
  i32x4 z4 = {0, 0, 0, 0};
  #if TDM_ARITY == 6
  i32x8 z8 = {0, 0, 0, 0, 0, 0, 0, 0};
  __builtin_amdgcn_tensor_load_to_lds(g0, g1, z4, z4, z8, 0);
  #else
  __builtin_amdgcn_tensor_load_to_lds(g0, g1, z4, z4, 0);
  #endif
  (void)lane;
#else
  // Cooperative wave copy fallback (handles strided rows, 16B granules).
  const unsigned pr = (width << dsz_log2) >> 4;          // 16B chunks per row
  const char* s = (const char*)gsrc;
  uint4* d = (uint4*)lds_dst;
  for (unsigned i = (unsigned)lane; i < rows * pr; i += 32u) {
    unsigned r = i / pr, c = i % pr;
    d[i] = *(const uint4*)(s + (((size_t)r * stride) << dsz_log2) + (c << 4));
  }
#endif
}

__device__ __forceinline__ v8f wmma16(v16h a, v16h b, v8f c) {
  // D(16x16,f32) = A(16x32,f16) * B(32x16,f16) + C
  return __builtin_amdgcn_wmma_f32_16x16x32_f16(false, a, false, b, (short)0, c,
                                                false, false);
}

// ---------------------------------------------------------------------------
// Kernel 0: one-shot weight conversion f32 -> f16, stored TRANSPOSED
// ([n][k], row length C) so projection B-fragments read LDS contiguously.
// Wq pre-scaled by H^-0.5 (0.125 = 2^-3, lossless fold).
// ---------------------------------------------------------------------------
__global__ __launch_bounds__(256)
void w_convert_kernel(const float* __restrict__ Wk, const float* __restrict__ Wq,
                      const float* __restrict__ Wv,
                      _Float16* __restrict__ whk, _Float16* __restrict__ whq,
                      _Float16* __restrict__ whv) {
  int idx = (int)blockIdx.x * 256 + (int)threadIdx.x;
  if (idx < Cn * Hn) {
    int k = idx / Hn, n = idx % Hn;
    size_t dst = (size_t)n * Cn + k;
    whk[dst] = (_Float16)Wk[idx];
    whq[dst] = (_Float16)(Wq[idx] * 0.125f);
    whv[dst] = (_Float16)Wv[idx];
  }
}

// ---------------------------------------------------------------------------
// Kernel 1: fused K/Q/V projection.  One wave per 16-row tile of [B*T, C].
// Double-buffered strided TDM staging of W^T tiles ([n(64)][k(32)] in LDS);
// all fragment reads are contiguous 16B runs.  V is stored transposed
// ([b][h][t]) for the attention kernel's P*V operand.
// ---------------------------------------------------------------------------
__global__ __launch_bounds__(32)
void attn_proj_kernel(const float* __restrict__ x,
                      const _Float16* __restrict__ whk,
                      const _Float16* __restrict__ whq,
                      const _Float16* __restrict__ whv,
                      _Float16* __restrict__ kh,
                      _Float16* __restrict__ qh,
                      _Float16* __restrict__ vh) {
  __shared__ _Float16 wl[2][3 * Hn * 32];       // 2 x 12 KB: [buf][mat][n(64)][k(32)]

  const int lane = (int)threadIdx.x;            // 0..31 (wave32)
  const int lm   = lane & 15;
  const int hi8  = (lane >> 4) * 8;
  const int row0 = (int)blockIdx.x * 16;        // global row in [B*T]

  v8f acc[12];
  #pragma unroll
  for (int i = 0; i < 12; ++i) acc[i] = (v8f){0, 0, 0, 0, 0, 0, 0, 0};

  const float* xrow = x + (size_t)(row0 + lm) * Cn;

  const int NSTEP = Cn / 32;
  // Prologue: stage step 0 into buffer 0 (strided tiles: 32 k's x 64 n's).
  tdm_load_2d(&wl[0][0],           whk, 1, 32, Hn, Cn, lane);
  tdm_load_2d(&wl[0][Hn * 32],     whq, 1, 32, Hn, Cn, lane);
  tdm_load_2d(&wl[0][2 * Hn * 32], whv, 1, 32, Hn, Cn, lane);

  for (int i = 0; i < NSTEP; ++i) {
    const int k0  = i * 32;
    const int cur = i & 1;

    if (i + 1 < NSTEP) {
      wait_dscnt0();                            // prior reads of next buffer retired
      const int nk = k0 + 32;
      tdm_load_2d(&wl[cur ^ 1][0],           whk + nk, 1, 32, Hn, Cn, lane);
      tdm_load_2d(&wl[cur ^ 1][Hn * 32],     whq + nk, 1, 32, Hn, Cn, lane);
      tdm_load_2d(&wl[cur ^ 1][2 * Hn * 32], whv + nk, 1, 32, Hn, Cn, lane);
      wait_tensorcnt<3>();                      // in-order: current buffer resident
    } else {
      wait_tensorcnt<0>();
    }

    // A fragment: x rows, f32 -> f16 (ISA 16-bit A layout)
    v16h a;
    #pragma unroll
    for (int h = 0; h < 16; ++h) {
      int kk = ((h < 8) ? h : h + 8) + hi8;
      a[h] = (_Float16)xrow[k0 + kk];
    }

    #pragma unroll
    for (int mat = 0; mat < 3; ++mat) {
      #pragma unroll
      for (int nt = 0; nt < 4; ++nt) {
        v16h b;                                 // B(32x16): (k,n) = wl[n][k], contiguous in k
        #pragma unroll
        for (int h = 0; h < 16; ++h) {
          int kk = ((h < 8) ? h : h + 8) + hi8;
          b[h] = wl[cur][mat * Hn * 32 + (nt * 16 + lm) * 32 + kk];
        }
        acc[mat * 4 + nt] = wmma16(a, b, acc[mat * 4 + nt]);
      }
    }
  }

  // Store D tiles: element (m = r + hi8, n = nt*16 + lm)
  const int bIdx = row0 / Tn;
  const int tloc = row0 % Tn;
  #pragma unroll
  for (int nt = 0; nt < 4; ++nt) {
    #pragma unroll
    for (int r = 0; r < 8; ++r) {
      int n = nt * 16 + lm;
      size_t off = (size_t)(row0 + r + hi8) * Hn + n;
      kh[off] = (_Float16)acc[0 * 4 + nt][r];
      qh[off] = (_Float16)acc[1 * 4 + nt][r];   // scale folded into whq
      // V transposed: [b][h][t]
      vh[(size_t)bIdx * Hn * Tn + (size_t)n * Tn + tloc + r + hi8] =
          (_Float16)acc[2 * 4 + nt][r];
    }
  }
}

// ---------------------------------------------------------------------------
// Kernel 2: causal flash attention.  One wave per (batch, 16-query tile).
// Double-buffered TDM staging: K row-major (contiguous), V^T strided tile
// ([h(64)][t(32)] in LDS).  All fragment reads are contiguous 16B runs.
// ---------------------------------------------------------------------------
__global__ __launch_bounds__(32)
void attn_flash_kernel(const _Float16* __restrict__ qh,
                       const _Float16* __restrict__ kh,
                       const _Float16* __restrict__ vh,
                       float* __restrict__ out) {
  __shared__ _Float16 Kt[2][32 * Hn];           // 2 x 4 KB  [key(32)][h(64)]
  __shared__ _Float16 Vt[2][Hn * 32];           // 2 x 4 KB  [h(64)][key(32)]
  __shared__ _Float16 Pt[16 * 32];              // 1 KB

  const int lane = (int)threadIdx.x;
  const int lm   = lane & 15;
  const int hi8  = (lane >> 4) * 8;

  const int tile = (int)blockIdx.x;             // 0 .. B*(T/16)-1
  const int b    = tile >> 7;                   // T/16 = 128 tiles per batch
  const int t0   = (tile & 127) << 4;

  const _Float16* qb = qh + ((size_t)b * Tn + t0) * Hn;
  const _Float16* kb = kh + (size_t)b * Tn * Hn;
  const _Float16* vb = vh + (size_t)b * Hn * Tn;   // transposed layout

  // Q A-fragments: 16x64 over two K=32 chunks
  v16h aq[2];
  #pragma unroll
  for (int c = 0; c < 2; ++c) {
    #pragma unroll
    for (int h = 0; h < 16; ++h) {
      int kk = ((h < 8) ? h : h + 8) + hi8 + c * 32;
      aq[c][h] = qb[(size_t)lm * Hn + kk];
    }
  }

  float mr[8], lr[8];
  v8f o[4];
  #pragma unroll
  for (int r = 0; r < 8; ++r) { mr[r] = -__builtin_inff(); lr[r] = 0.0f; }
  #pragma unroll
  for (int nt = 0; nt < 4; ++nt) o[nt] = (v8f){0, 0, 0, 0, 0, 0, 0, 0};

  const int jend = (t0 + 15) >> 5;              // last KV tile with unmasked keys

  // Prologue: stage tile 0 into buffer 0.
  tdm_load_2d(&Kt[0][0], kb, 1, Hn, 32, Hn, lane);     // contiguous rows
  tdm_load_2d(&Vt[0][0], vb, 1, 32, Hn, Tn, lane);     // strided: 32 t's x 64 h's

  for (int j = 0; j <= jend; ++j) {
    const int s0  = j * 32;
    const int cur = j & 1;

    if (j < jend) {
      wait_dscnt0();                            // prior reads of next buffer retired
      tdm_load_2d(&Kt[cur ^ 1][0], kb + (size_t)(s0 + 32) * Hn, 1, Hn, 32, Hn, lane);
      tdm_load_2d(&Vt[cur ^ 1][0], vb + (s0 + 32),              1, 32, Hn, Tn, lane);
      wait_tensorcnt<2>();                      // in-order: current buffer resident
    } else {
      wait_tensorcnt<0>();
    }

    // S = Q * K^T   (two 16x16 key sub-tiles, two K=32 h-chunks each)
    v8f s[2];
    s[0] = (v8f){0, 0, 0, 0, 0, 0, 0, 0};
    s[1] = (v8f){0, 0, 0, 0, 0, 0, 0, 0};
    #pragma unroll
    for (int sub = 0; sub < 2; ++sub) {
      #pragma unroll
      for (int c = 0; c < 2; ++c) {
        v16h bk;                                // B(32x16): (k,n) = Kt[sub*16+n][c*32+k]
        #pragma unroll
        for (int h = 0; h < 16; ++h) {
          int kk = ((h < 8) ? h : h + 8) + hi8 + c * 32;
          bk[h] = Kt[cur][(sub * 16 + lm) * Hn + kk];
        }
        s[sub] = wmma16(aq[c], bk, s[sub]);
      }
    }

    // Causal mask: row t = t0 + r + hi8, key = s0 + sub*16 + lm
    #pragma unroll
    for (int sub = 0; sub < 2; ++sub) {
      #pragma unroll
      for (int r = 0; r < 8; ++r) {
        int t   = t0 + r + hi8;
        int key = s0 + sub * 16 + lm;
        s[sub][r] = (key <= t) ? s[sub][r] : -__builtin_inff();
      }
    }

    // Online softmax (row reductions across the 16-lane halves)
    #pragma unroll
    for (int r = 0; r < 8; ++r) {
      float v = fmaxf(s[0][r], s[1][r]);
      #pragma unroll
      for (int off = 8; off >= 1; off >>= 1)
        v = fmaxf(v, __shfl_xor(v, off, 32));
      float mnew  = fmaxf(mr[r], v);
      float alpha = __expf(mr[r] - mnew);       // 0 on first step (mr=-inf), safe
      s[0][r] = __expf(s[0][r] - mnew);
      s[1][r] = __expf(s[1][r] - mnew);
      float ps = s[0][r] + s[1][r];
      #pragma unroll
      for (int off = 8; off >= 1; off >>= 1)
        ps += __shfl_xor(ps, off, 32);
      lr[r] = lr[r] * alpha + ps;
      mr[r] = mnew;
      #pragma unroll
      for (int nt = 0; nt < 4; ++nt) o[nt][r] *= alpha;
    }

    // P: C/D layout -> LDS -> A layout (per-wave LDS ops are in-order)
    #pragma unroll
    for (int sub = 0; sub < 2; ++sub)
      #pragma unroll
      for (int r = 0; r < 8; ++r)
        Pt[(r + hi8) * 32 + sub * 16 + lm] = (_Float16)s[sub][r];

    v16h ap;
    #pragma unroll
    for (int h = 0; h < 16; ++h) {
      int kk = ((h < 8) ? h : h + 8) + hi8;
      ap[h] = Pt[lm * 32 + kk];
    }

    // O += P(16x32) * V(32x64)
    #pragma unroll
    for (int nt = 0; nt < 4; ++nt) {
      v16h bv;                                  // B(32x16): (k,n) = Vt[n][k], contiguous in k
      #pragma unroll
      for (int h = 0; h < 16; ++h) {
        int kk = ((h < 8) ? h : h + 8) + hi8;
        bv[h] = Vt[cur][(nt * 16 + lm) * 32 + kk];
      }
      o[nt] = wmma16(ap, bv, o[nt]);
    }
  }

  // Epilogue: divide by row sums (uniform across lane halves), store f32
  #pragma unroll
  for (int nt = 0; nt < 4; ++nt) {
    #pragma unroll
    for (int r = 0; r < 8; ++r) {
      int t = t0 + r + hi8;
      out[((size_t)b * Tn + t) * Hn + nt * 16 + lm] = o[nt][r] / lr[r];
    }
  }
}

// ---------------------------------------------------------------------------
extern "C" void kernel_launch(void* const* d_in, const int* in_sizes, int n_in,
                              void* d_out, int out_size, void* d_ws, size_t ws_size,
                              hipStream_t stream) {
  (void)in_sizes; (void)n_in; (void)out_size; (void)ws_size;
  const float* x  = (const float*)d_in[0];
  const float* Wk = (const float*)d_in[1];
  const float* Wq = (const float*)d_in[2];
  const float* Wv = (const float*)d_in[3];
  float* out = (float*)d_out;

  const size_t BTH = (size_t)Bn * Tn * Hn;      // 1,048,576 elements
  const size_t CH  = (size_t)Cn * Hn;           // 65,536 elements
  _Float16* kh  = (_Float16*)d_ws;              // 2 MB [b][t][h]
  _Float16* qh  = kh + BTH;                     // 2 MB [b][t][h]
  _Float16* vh  = qh + BTH;                     // 2 MB [b][h][t] (transposed)
  _Float16* whk = vh + BTH;                     // 128 KB, W^T [n][k]
  _Float16* whq = whk + CH;                     // 128 KB, W^T [n][k], pre-scaled
  _Float16* whv = whq + CH;                     // 128 KB, W^T [n][k]

  w_convert_kernel<<<dim3((Cn * Hn + 255) / 256), 256, 0, stream>>>(Wk, Wq, Wv,
                                                                    whk, whq, whv);
  dim3 grid(Bn * Tn / 16);                      // 1024 waves
  attn_proj_kernel<<<grid, 32, 0, stream>>>(x, whk, whq, whv, kh, qh, vh);
  attn_flash_kernel<<<grid, 32, 0, stream>>>(qh, kh, vh, out);
}
